// CHD_GNN_77738908058358
// MI455X (gfx1250) — matrix operand order repaired and checked by reading, not compiled, not run on hardware
//
#include <hip/hip_runtime.h>

typedef __attribute__((ext_vector_type(16))) _Float16 v16h;
typedef __attribute__((ext_vector_type(8)))  float    v8f;

#define DEVINL __device__ __forceinline__

// Monotone order-preserving f32 -> u32 encoding (for atomicMin/Max on floats)
DEVINL unsigned encF(float f) {
  int b = __float_as_int(f);
  return (b >= 0) ? ((unsigned)b | 0x80000000u) : (unsigned)(~b);
}
DEVINL float decF(unsigned u) {
  int b = (u & 0x80000000u) ? (int)(u & 0x7FFFFFFFu) : (int)(~u);
  return __int_as_float(b);
}

__global__ void k_zero_int(int* p, int n) {
  int i = blockIdx.x * blockDim.x + threadIdx.x;
  if (i < n) p[i] = 0;
}

__global__ void k_count_deg(const int* __restrict__ dst, int* __restrict__ cnt, int E) {
  int i = blockIdx.x * blockDim.x + threadIdx.x;
  if (i < E) atomicAdd(&cnt[dst[i]], 1);
}

__global__ void k_init_agg(float* __restrict__ S, float* __restrict__ Q,
                           unsigned* __restrict__ Mn, unsigned* __restrict__ Mx, int n) {
  int i = blockIdx.x * blockDim.x + threadIdx.x;
  if (i < n) {
    S[i] = 0.f; Q[i] = 0.f;
    Mn[i] = 0xFF800000u;  // enc(+inf)
    Mx[i] = 0x007FFFFFu;  // enc(-inf)
  }
}

// One thread per (edge, tower). Computes per-tower pre-MLP message (2F->F) and
// scatters into sum / sumsq / min / max planes with L2 atomics.
template <int F>
__global__ void edge_msg(const int* __restrict__ src, const int* __restrict__ dst,
                         const float* __restrict__ X, const float* __restrict__ Wpre,
                         const float* __restrict__ bpre, float* __restrict__ S,
                         float* __restrict__ Q, unsigned* __restrict__ Mn,
                         unsigned* __restrict__ Mx, int E, int T, int Cin) {
  int id = blockIdx.x * blockDim.x + threadIdx.x;
  int tot = E * T;
  if (id >= tot) return;
  int e = id / T, t = id - e * T;
  int s = src[e], d = dst[e];
  float xd[F], xs[F];
#pragma unroll
  for (int f = 0; f < F; ++f) {
    xd[f] = X[(size_t)d * Cin + t * F + f];
    xs[f] = X[(size_t)s * Cin + t * F + f];
  }
  const float* W = Wpre + (size_t)t * (2 * F) * F;
#pragma unroll
  for (int g = 0; g < F; ++g) {
    float m = bpre[t * F + g];
#pragma unroll
    for (int f = 0; f < F; ++f)
      m += xd[f] * W[f * F + g] + xs[f] * W[(F + f) * F + g];
    int idx = d * Cin + t * F + g;
    atomicAdd(&S[idx], m);
    atomicAdd(&Q[idx], m * m);
    atomicMin(&Mn[idx], encF(m));
    atomicMax(&Mx[idx], encF(m));
  }
}

// One thread per (node, tower): finalize aggregators, apply 3 degree scalers,
// build the 13F vector, run the per-tower post-MLP (13F -> Fout).
template <int F>
__global__ void node_post(const float* __restrict__ X, const float* __restrict__ S,
                          const float* __restrict__ Q, const unsigned* __restrict__ Mn,
                          const unsigned* __restrict__ Mx, const int* __restrict__ cnt,
                          const float* __restrict__ Wpost, const float* __restrict__ bpost,
                          float* __restrict__ out, int N, int T, int Fout,
                          int Cin, int Cout, float adl) {
  int id = blockIdx.x * blockDim.x + threadIdx.x;
  int tot = N * T;
  if (id >= tot) return;
  int n = id / T, t = id - n * T;
  int c = cnt[n];
  float deg = (c > 1) ? (float)c : 1.0f;
  float ld = logf(deg + 1.0f);
  float amp = ld / adl, att = adl / ld;
  float v[13 * F];
#pragma unroll
  for (int f = 0; f < F; ++f) {
    int idx = n * Cin + t * F + f;
    float s = S[idx], q = Q[idx];
    float mean = s / deg;
    float sd = sqrtf(fmaxf(q / deg - mean * mean, 1e-5f));
    float mn = (c > 0) ? decF(Mn[idx]) : 0.f;
    float mx = (c > 0) ? decF(Mx[idx]) : 0.f;
    v[f] = X[idx];
    v[1 * F + f] = mean;       v[2 * F + f] = mn;       v[3 * F + f] = mx;       v[4 * F + f] = sd;
    v[5 * F + f] = mean * amp; v[6 * F + f] = mn * amp; v[7 * F + f] = mx * amp; v[8 * F + f] = sd * amp;
    v[9 * F + f] = mean * att; v[10 * F + f] = mn * att; v[11 * F + f] = mx * att; v[12 * F + f] = sd * att;
  }
  const float* W = Wpost + (size_t)t * (13 * F) * Fout;
  for (int g = 0; g < Fout; ++g) {
    float acc = bpost[t * Fout + g];
#pragma unroll
    for (int f = 0; f < 13 * F; ++f) acc += v[f] * W[f * Fout + g];
    out[(size_t)n * Cout + t * Fout + g] = acc;
  }
}

// Dense H = A(N,C) @ W(C,C) + b via V_WMMA_F32_16X16X32_F16, templated on C so
// all K-guards are compile-time and fragment loads are vectorized + branchless.
// grid = (N/64, C/16); 128 threads = 4 waves; one 16x16 output tile per wave.
// W_lin is staged once per workgroup into LDS as f16, transposed (Wt[n][k]),
// K-padded to 32 so C==16 needs no lane predication.
template <int C>
__global__ void lin_wmma_t(const float* __restrict__ A, const float* __restrict__ W,
                           const float* __restrict__ bias, float* __restrict__ H,
                           int N) {
  constexpr int KP = (C < 32) ? 32 : C;  // padded K extent
  __shared__ _Float16 Wt[C * KP];

  int tid = threadIdx.x;
  // Cooperative transpose + f32->f16 convert + zero-pad of W (C*C <= 4096 elems)
  for (int idx = tid; idx < C * KP; idx += 128) {
    int n = idx / KP, k = idx - n * KP;
    Wt[idx] = (k < C) ? (_Float16)W[k * C + n] : (_Float16)0.f;
  }
  __syncthreads();

  int lane = tid & 31;
  int wave = tid >> 5;
  int row0 = (blockIdx.x * 4 + wave) * 16;
  int colT = blockIdx.y * 16;
  int hi = lane >> 4;   // half-wave selector
  int lm = lane & 15;
  const float* Ap = A + (size_t)(row0 + lm) * C;  // A: lane holds row M = lane&15

  v8f acc = {};
#pragma unroll
  for (int kk = 0; kk < C; kk += 32) {
    // A 16x32 f16 layout: lanes 0-15 hold K = kk+[0..7], kk+[16..23];
    //                     lanes 16-31 hold K = kk+[8..15], kk+[24..31].
    int k0 = kk + hi * 8;
    float4 p0 = *(const float4*)(Ap + k0);
    float4 p1 = *(const float4*)(Ap + k0 + 4);
    float4 p2, p3;
    if constexpr (C >= 32) {
      p2 = *(const float4*)(Ap + k0 + 16);
      p3 = *(const float4*)(Ap + k0 + 20);
    } else {  // C == 16: K = 16..31 is compile-time zero padding
      p2 = make_float4(0.f, 0.f, 0.f, 0.f);
      p3 = make_float4(0.f, 0.f, 0.f, 0.f);
    }
    v16h a;
    a[0] = (_Float16)p0.x;  a[1] = (_Float16)p0.y;
    a[2] = (_Float16)p0.z;  a[3] = (_Float16)p0.w;
    a[4] = (_Float16)p1.x;  a[5] = (_Float16)p1.y;
    a[6] = (_Float16)p1.z;  a[7] = (_Float16)p1.w;
    a[8] = (_Float16)p2.x;  a[9] = (_Float16)p2.y;
    a[10] = (_Float16)p2.z; a[11] = (_Float16)p2.w;
    a[12] = (_Float16)p3.x; a[13] = (_Float16)p3.y;
    a[14] = (_Float16)p3.z; a[15] = (_Float16)p3.w;

    // B 32x16 f16 layout: lane = column; lanes 0-15 K=kk+[0..7],kk+[16..23],
    // lanes 16-31 K=kk+[8..15],kk+[24..31] (matches A) in transposed LDS image.
    const _Float16* Bp = &Wt[(colT + lm) * KP + kk + hi * 8];
    v16h b;
#pragma unroll
    for (int s = 0; s < 8; ++s) { b[s] = Bp[s]; b[s + 8] = Bp[s + 16]; }

    acc = __builtin_amdgcn_wmma_f32_16x16x32_f16(false, a, false, b, (short)0, acc,
                                                 false, false);
  }

  float bv = bias[colT + lm];
#pragma unroll
  for (int r = 0; r < 8; ++r) {
    int row = row0 + r + hi * 8;  // C/D: VGPR r holds M=r (lanes 0-15), M=r+8 (lanes 16-31)
    H[(size_t)row * C + colT + lm] = acc[r] + bv;
  }
}

// Fallback GEMM for C < 16 (c_out == 8 blocks and the final 8x8 head).
__global__ void lin_scalar(const float* __restrict__ A, const float* __restrict__ W,
                           const float* __restrict__ b, float* __restrict__ H,
                           int N, int Cin, int Cout) {
  int id = blockIdx.x * blockDim.x + threadIdx.x;
  int tot = N * Cout;
  if (id >= tot) return;
  int n = id / Cout, g = id - n * Cout;
  float acc = b[g];
  for (int k = 0; k < Cin; ++k) acc += A[(size_t)n * Cin + k] * W[k * Cout + g];
  H[id] = acc;
}

// Per-channel mean/var over all N rows (training-mode BatchNorm, biased var).
__global__ void bn_stats(const float* __restrict__ H, int N, int C, float* __restrict__ stats) {
  int ch = blockIdx.x;
  float s = 0.f, q = 0.f;
  for (int n = threadIdx.x; n < N; n += blockDim.x) {
    float v = H[(size_t)n * C + ch];
    s += v; q += v * v;
  }
  __shared__ float ss[256];
  __shared__ float sq[256];
  int tid = threadIdx.x;
  ss[tid] = s; sq[tid] = q;
  __syncthreads();
  for (int w = 128; w > 0; w >>= 1) {
    if (tid < w) { ss[tid] += ss[tid + w]; sq[tid] += sq[tid + w]; }
    __syncthreads();
  }
  if (tid == 0) {
    float mu = ss[0] / (float)N;
    stats[ch] = mu;
    stats[C + ch] = sq[0] / (float)N - mu * mu;
  }
}

// y = relu(gamma*(h-mu)*rsqrt(var+eps)+beta), written at channel offset `off`
// with row stride ldOut (implements the U-Net channel concats in place).
__global__ void bn_apply_relu(const float* __restrict__ H, const float* __restrict__ stats,
                              const float* __restrict__ gamma, const float* __restrict__ beta,
                              float* __restrict__ out, int N, int C, int ldOut, int off) {
  int id = blockIdx.x * blockDim.x + threadIdx.x;
  int tot = N * C;
  if (id >= tot) return;
  int n = id / C, ch = id - n * C;
  float mu = stats[ch], var = stats[C + ch];
  float v = gamma[ch] * (H[id] - mu) * rsqrtf(var + 1e-5f) + beta[ch];
  out[(size_t)n * ldOut + off + ch] = fmaxf(v, 0.f);
}

__global__ void copy_ch(const float* __restrict__ srcB, int ldS, float* __restrict__ dstB,
                        int ldD, int off, int C, int N) {
  int id = blockIdx.x * blockDim.x + threadIdx.x;
  int tot = N * C;
  if (id >= tot) return;
  int n = id / C, ch = id - n * C;
  dstB[(size_t)n * ldD + off + ch] = srcB[(size_t)n * ldS + ch];
}

extern "C" void kernel_launch(void* const* d_in, const int* in_sizes, int n_in,
                              void* d_out, int out_size, void* d_ws, size_t ws_size,
                              hipStream_t stream) {
  const int N = 131072, E = 524288;
  const float ADL = 1.6094379124341003f;  // log(5)

  const float* x = (const float*)d_in[0];
  const int* ei = (const int*)d_in[1];
  const int* src = ei;
  const int* dst = ei + E;

  // Per-block params in setup_inputs insertion order:
  // W_pre, b_pre, W_post, b_post, W_lin, b_lin, gamma, beta
  const float* P[14][8];
  for (int i = 0; i < 14; ++i)
    for (int j = 0; j < 8; ++j) P[i][j] = (const float*)d_in[2 + 8 * i + j];
  const float* W_out = (const float*)d_in[2 + 112];
  const float* b_out = (const float*)d_in[3 + 112];

  // Workspace layout
  float* ws = (float*)d_ws;
  int* cnt = (int*)ws;
  float* featA = ws + (size_t)N;
  float* featB = featA + (size_t)N * 96;
  float* skip1 = featB + (size_t)N * 96;
  float* skip2 = skip1 + (size_t)N * 8;
  float* skip3 = skip2 + (size_t)N * 16;
  float* aggS = skip3 + (size_t)N * 32;
  float* aggQ = aggS + (size_t)N * 96;
  unsigned* aggMn = (unsigned*)(aggQ + (size_t)N * 96);
  unsigned* aggMx = aggMn + (size_t)N * 96;
  float* post = (float*)(aggMx + (size_t)N * 96);
  float* hbuf = post + (size_t)N * 64;
  float* stats = hbuf + (size_t)N * 64;

  auto cdiv = [](long a, long b) { return (unsigned)((a + b - 1) / b); };

  // In-degree (shared by all blocks)
  k_zero_int<<<cdiv(N, 256), 256, 0, stream>>>(cnt, N);
  k_count_deg<<<cdiv(E, 256), 256, 0, stream>>>(dst, cnt, E);

  const int CI[14] = {1, 8, 8, 16, 16, 32, 32, 64, 96, 64, 48, 32, 24, 16};
  const int CO[14] = {8, 8, 16, 16, 32, 32, 64, 64, 64, 32, 32, 16, 16, 8};

  // Routing: input buffer, output buffer, output row stride, output channel offset
  const float* inP[14] = {x,     featA, skip1, featA, skip2, featA, skip3,
                          featA, featB, featA, featB, featA, featB, featA};
  float* outP[14] = {featA, skip1, featA, skip2, featA, skip3, featA,
                     featB, featA, featB, featA, featB, featA, featB};
  const int ldO[14] = {8, 8, 16, 16, 32, 32, 64, 96, 64, 48, 32, 24, 16, 8};
  const int offO[14] = {0, 0, 0, 0, 0, 0, 0, 32, 0, 16, 0, 8, 0, 0};

  for (int i = 0; i < 14; ++i) {
    int ci = CI[i], co = CO[i];
    int t = (ci < co) ? ci : co;
    if (ci % t) t >>= 1;
    int F = ci / t, Fo = co / t;

    k_init_agg<<<cdiv((long)N * ci, 256), 256, 0, stream>>>(aggS, aggQ, aggMn, aggMx, N * ci);

    unsigned ge = cdiv((long)E * t, 256);
    unsigned gn = cdiv((long)N * t, 256);
    switch (F) {
      case 1:
        edge_msg<1><<<ge, 256, 0, stream>>>(src, dst, inP[i], P[i][0], P[i][1],
                                            aggS, aggQ, aggMn, aggMx, E, t, ci);
        node_post<1><<<gn, 256, 0, stream>>>(inP[i], aggS, aggQ, aggMn, aggMx, cnt,
                                             P[i][2], P[i][3], post, N, t, Fo, ci, co, ADL);
        break;
      case 2:
        edge_msg<2><<<ge, 256, 0, stream>>>(src, dst, inP[i], P[i][0], P[i][1],
                                            aggS, aggQ, aggMn, aggMx, E, t, ci);
        node_post<2><<<gn, 256, 0, stream>>>(inP[i], aggS, aggQ, aggMn, aggMx, cnt,
                                             P[i][2], P[i][3], post, N, t, Fo, ci, co, ADL);
        break;
      default:
        edge_msg<3><<<ge, 256, 0, stream>>>(src, dst, inP[i], P[i][0], P[i][1],
                                            aggS, aggQ, aggMn, aggMx, E, t, ci);
        node_post<3><<<gn, 256, 0, stream>>>(inP[i], aggS, aggQ, aggMn, aggMx, cnt,
                                             P[i][2], P[i][3], post, N, t, Fo, ci, co, ADL);
        break;
    }

    if (co >= 16) {
      dim3 g(N / 64, co / 16);
      if (co == 16)
        lin_wmma_t<16><<<g, 128, 0, stream>>>(post, P[i][4], P[i][5], hbuf, N);
      else if (co == 32)
        lin_wmma_t<32><<<g, 128, 0, stream>>>(post, P[i][4], P[i][5], hbuf, N);
      else
        lin_wmma_t<64><<<g, 128, 0, stream>>>(post, P[i][4], P[i][5], hbuf, N);
    } else {
      lin_scalar<<<cdiv((long)N * co, 256), 256, 0, stream>>>(post, P[i][4], P[i][5],
                                                              hbuf, N, co, co);
    }

    bn_stats<<<co, 256, 0, stream>>>(hbuf, N, co, stats);
    bn_apply_relu<<<cdiv((long)N * co, 256), 256, 0, stream>>>(
        hbuf, stats, P[i][6], P[i][7], outP[i], N, co, ldO[i], offO[i]);

    // Skip concatenation for the decoder inputs
    if (i == 7)
      copy_ch<<<cdiv((long)N * 32, 256), 256, 0, stream>>>(skip3, 32, featB, 96, 0, 32, N);
    else if (i == 9)
      copy_ch<<<cdiv((long)N * 16, 256), 256, 0, stream>>>(skip2, 16, featB, 48, 0, 16, N);
    else if (i == 11)
      copy_ch<<<cdiv((long)N * 8, 256), 256, 0, stream>>>(skip1, 8, featB, 24, 0, 8, N);
  }

  // Final head: out = h @ W_out + b_out  (N x 8)
  lin_scalar<<<cdiv((long)N * 8, 256), 256, 0, stream>>>(featB, W_out, b_out,
                                                         (float*)d_out, N, 8, 8);
}
